// ContextGuidedSelectiveStateModel_82987358094113
// MI455X (gfx1250) — matrix operand order repaired and checked by reading, not compiled, not run on hardware
//
#include <hip/hip_runtime.h>
#include <hip/hip_bf16.h>
#include <math.h>

// ---------------------------------------------------------------------------
// Types for CDNA5 WMMA (gfx1250, wave32)
// ---------------------------------------------------------------------------
typedef _Float16     f16x16 __attribute__((ext_vector_type(16)));
typedef float        f32x8  __attribute__((ext_vector_type(8)));
typedef unsigned int u32x8  __attribute__((ext_vector_type(8)));
typedef int          v4i    __attribute__((ext_vector_type(4)));

#define GLOBAL_AS __attribute__((address_space(1)))
#define LDS_AS    __attribute__((address_space(3)))

#if defined(__AMDGCN__) && __has_builtin(__builtin_amdgcn_global_load_async_to_lds_b128) && __has_builtin(__builtin_amdgcn_s_wait_asynccnt)
#define USE_ASYNC_LDS 1
#else
#define USE_ASYNC_LDS 0
#endif

// Split an f32 into hi/lo f16: x ~= (float)h + (float)l (~2^-22 rel err).
__device__ __forceinline__ void split2(float x, unsigned short& h, unsigned short& l) {
    _Float16 hh = (_Float16)x;
    float    r  = x - (float)hh;
    _Float16 ll = (_Float16)r;
    h = __builtin_bit_cast(unsigned short, hh);
    l = __builtin_bit_cast(unsigned short, ll);
}

__device__ __forceinline__ f16x16 frag_from(uint4 p0, uint4 p1) {
    u32x8 w;
    w[0] = p0.x; w[1] = p0.y; w[2] = p0.z; w[3] = p0.w;
    w[4] = p1.x; w[5] = p1.y; w[6] = p1.z; w[7] = p1.w;
    return __builtin_bit_cast(f16x16, w);
}

__device__ __forceinline__ float sigmoidf_(float x) { return 1.0f / (1.0f + __expf(-x)); }
__device__ __forceinline__ float siluf_(float x)    { return x * sigmoidf_(x); }

// ---------------------------------------------------------------------------
// Model constants
// ---------------------------------------------------------------------------
#define EMB   768
#define INNER 1536
#define DRANK 48
#define STATE 16
#define SEQL  2048
#define NTOK  4096          // B * L
#define XPC   80            // DRANK + 2*STATE

// ---------------------------------------------------------------------------
// Prep: split weight W[K,N] (f32, row-major) into transposed hi/lo f16 planes
// Bh/Bl[Np][Kp] (ushort), zero-padded so GEMM tiles need no bounds checks.
// ---------------------------------------------------------------------------
__global__ __launch_bounds__(256)
void split_weight_kernel(const float* __restrict__ W, int K, int N, int Kp, int Np,
                         unsigned short* __restrict__ Bh, unsigned short* __restrict__ Bl) {
    const size_t idx = (size_t)blockIdx.x * 256 + threadIdx.x;
    if (idx >= (size_t)Np * Kp) return;
    const int n = (int)(idx / Kp);
    const int k = (int)(idx % Kp);
    float v = (n < N && k < K) ? W[(size_t)k * N + n] : 0.f;
    unsigned short h, l;
    split2(v, h, l);
    Bh[idx] = h;
    Bl[idx] = l;
}

// ---------------------------------------------------------------------------
// Kernel 1: context layernorm + drive/gbias projections (tiny)
// ---------------------------------------------------------------------------
__global__ __launch_bounds__(256)
void ctx_kernel(const float* __restrict__ rctx,
                const float* __restrict__ ln_g, const float* __restrict__ ln_b,
                const float* __restrict__ W_scale, const float* __restrict__ b_scale,
                const float* __restrict__ W_gbias, const float* __restrict__ b_gbias,
                float* __restrict__ drive, float* __restrict__ gbias_out) {
    __shared__ float red[256];
    __shared__ float cnorm[EMB];
    const int b = blockIdx.x;
    const int tid = threadIdx.x;
    const float* row = rctx + (size_t)b * EMB;

    float s = 0.f;
    for (int k = tid; k < EMB; k += 256) s += row[k];
    red[tid] = s; __syncthreads();
    for (int off = 128; off; off >>= 1) { if (tid < off) red[tid] += red[tid + off]; __syncthreads(); }
    const float mean = red[0] * (1.0f / EMB);
    __syncthreads();

    float s2 = 0.f;
    for (int k = tid; k < EMB; k += 256) { float d = row[k] - mean; s2 += d * d; }
    red[tid] = s2; __syncthreads();
    for (int off = 128; off; off >>= 1) { if (tid < off) red[tid] += red[tid + off]; __syncthreads(); }
    const float var = red[0] * (1.0f / EMB);
    const float inv = rsqrtf(var + 1e-5f);
    __syncthreads();

    for (int k = tid; k < EMB; k += 256)
        cnorm[k] = (row[k] - mean) * inv * ln_g[k] + ln_b[k];
    __syncthreads();

    for (int j = tid; j < INNER; j += 256) {
        float a1 = b_scale[j];
        float a2 = b_gbias[j];
        for (int k = 0; k < EMB; ++k) {
            float v = cnorm[k];
            a1 += v * W_scale[(size_t)k * INNER + j];
            a2 += v * W_gbias[(size_t)k * INNER + j];
        }
        drive[b * INNER + j]     = sigmoidf_(a1);
        gbias_out[b * INNER + j] = a2;
    }
}

// ---------------------------------------------------------------------------
// WMMA GEMM:  out[M,N] = epilogue( A[M,Kp] @ B[Kp,N] + bias[N] )
// A: f32 activations, split to hi/lo f16 in-kernel (unguarded float2 loads).
// B: pre-split transposed planes BhT/BlT[Np][Kp]; tiles staged to LDS with
//    GLOBAL_LOAD_ASYNC_TO_LDS_B128 (ASYNCcnt) when available.
// Each fragment is exactly 2x ds_read_b128. 128x64 block tile, 8 waves,
// 2x2 16x16 tiles/wave; 3 WMMAs per tile pair for f32-accurate emulation.
// EPI: 0 = plain (+bias), 1 = split x/z with drive/gbias, 2 = softplus(+bias)
// ---------------------------------------------------------------------------
template <int EPI>
__global__ __launch_bounds__(256)
void wmma_gemm_kernel(const float* __restrict__ A, int lda,
                      const unsigned short* __restrict__ BhT,
                      const unsigned short* __restrict__ BlT,
                      int Kp,                            // padded K (mult of 32) = plane row stride
                      const float* __restrict__ bias,
                      float* __restrict__ out, int ldo,
                      int N,
                      const float* __restrict__ aux1,    // drive  (EPI 1)
                      const float* __restrict__ aux2,    // gbias  (EPI 1)
                      float* __restrict__ out2) {        // z-out  (EPI 1)
    // A planes: u32 words, word kw holds f16(k=2kw),f16(k=2kw+1). Row stride 20
    // words -> 16B-aligned rows, distinct banks across 16 consecutive rows.
    __shared__ unsigned int   sAh[128][20];
    __shared__ unsigned int   sAl[128][20];
    // B planes: transposed [n][k] ushort; row stride 40 -> 16B-aligned rows.
    __shared__ unsigned short sBh[64][40];
    __shared__ unsigned short sBl[64][40];

    const int tid  = threadIdx.x;
    const int lane = tid & 31;
    const int wid  = tid >> 5;
    const int wm   = (wid & 3) * 32;   // wave row offset in block tile
    const int wn   = (wid >> 2) * 32;  // wave col offset in block tile
    const int bm   = blockIdx.x * 128;
    const int bn   = blockIdx.y * 64;

    const int half = lane >> 4;        // K half select (ISA 7.12.2 layouts)
    const int l16  = lane & 15;

    f32x8 acc[2][2];
#pragma unroll
    for (int mi = 0; mi < 2; ++mi)
#pragma unroll
        for (int ni = 0; ni < 2; ++ni) {
            f32x8 z = {0.f, 0.f, 0.f, 0.f, 0.f, 0.f, 0.f, 0.f};
            acc[mi][ni] = z;
        }

    for (int k0 = 0; k0 < Kp; k0 += 32) {
        // --- stage B tile (32 x 64) from pre-split planes: pure byte copies ---
        // 64 rows x 64B x 2 planes = 512 x 16B chunks, 2 per thread.
#pragma unroll
        for (int i = 0; i < 2; ++i) {
            const int idx = i * 256 + tid;
            const int nn  = (idx >> 2) & 63;
            const int c4  = idx & 3;
            const bool lo = idx >= 256;
            const unsigned short* g = (lo ? BlT : BhT) + (size_t)(bn + nn) * Kp + k0 + c4 * 8;
            unsigned short*       d = lo ? &sBl[nn][c4 * 8] : &sBh[nn][c4 * 8];
#if USE_ASYNC_LDS
            __builtin_amdgcn_global_load_async_to_lds_b128(
                (GLOBAL_AS v4i*)(unsigned short*)g,
                (LDS_AS v4i*)d, 0, 0);
#else
            *(uint4*)d = *(const uint4*)g;
#endif
        }
        // --- stage A tile (128 x 32): 2048 words/plane, 8 per thread ---
#pragma unroll
        for (int i = 0; i < 8; ++i) {
            const int idx = i * 256 + tid;
            const int r   = idx >> 4;          // row 0..127
            const int kw  = idx & 15;          // word col 0..15
            const float2 v = *(const float2*)(A + (size_t)(bm + r) * lda + k0 + kw * 2);
            unsigned short h0, l0, h1, l1;
            split2(v.x, h0, l0);
            split2(v.y, h1, l1);
            sAh[r][kw] = (unsigned int)h0 | ((unsigned int)h1 << 16);
            sAl[r][kw] = (unsigned int)l0 | ((unsigned int)l1 << 16);
        }
        // prefetch next A K-tile (global_prefetch_b8)
        if (k0 + 32 < Kp)
            __builtin_prefetch(&A[(size_t)(bm + (tid >> 1)) * lda + (k0 + 32) + (tid & 1) * 16], 0, 1);
#if USE_ASYNC_LDS
        __builtin_amdgcn_s_wait_asynccnt(0);
#endif
        __syncthreads();

        // --- fragments: pure ds_read_b128, no shuffles ---
        f16x16 aH[2], aL[2], bH[2], bL[2];
#pragma unroll
        for (int mi = 0; mi < 2; ++mi) {
            const unsigned int* rH = &sAh[wm + mi * 16 + l16][0];
            const unsigned int* rL = &sAl[wm + mi * 16 + l16][0];
            aH[mi] = frag_from(*(const uint4*)(rH + 8 * half), *(const uint4*)(rH + 4 + 8 * half));
            aL[mi] = frag_from(*(const uint4*)(rL + 8 * half), *(const uint4*)(rL + 4 + 8 * half));
        }
#pragma unroll
        for (int ni = 0; ni < 2; ++ni) {
            const uint4* cH = (const uint4*)&sBh[wn + ni * 16 + l16][16 * half];
            const uint4* cL = (const uint4*)&sBl[wn + ni * 16 + l16][16 * half];
            bH[ni] = frag_from(cH[0], cH[1]);
            bL[ni] = frag_from(cL[0], cL[1]);
        }

        // --- 3-product f32 emulation: hi*hi + hi*lo + lo*hi ---
#pragma unroll
        for (int mi = 0; mi < 2; ++mi)
#pragma unroll
            for (int ni = 0; ni < 2; ++ni) {
                acc[mi][ni] = __builtin_amdgcn_wmma_f32_16x16x32_f16(
                    false, aH[mi], false, bH[ni], (short)0, acc[mi][ni], false, false);
                acc[mi][ni] = __builtin_amdgcn_wmma_f32_16x16x32_f16(
                    false, aH[mi], false, bL[ni], (short)0, acc[mi][ni], false, false);
                acc[mi][ni] = __builtin_amdgcn_wmma_f32_16x16x32_f16(
                    false, aL[mi], false, bH[ni], (short)0, acc[mi][ni], false, false);
            }
        __syncthreads();
    }

    // --- epilogue + store (C/D layout: n = lane&15, row = r + 8*(lane>>4)) ---
#pragma unroll
    for (int mi = 0; mi < 2; ++mi)
#pragma unroll
        for (int ni = 0; ni < 2; ++ni) {
            const int gn = bn + wn + ni * 16 + l16;
            if (gn >= N) continue;
            const float bv = bias ? bias[gn] : 0.f;
#pragma unroll
            for (int r = 0; r < 8; ++r) {
                const int gm = bm + wm + mi * 16 + r + 8 * half;
                float v = acc[mi][ni][r] + bv;
                if (EPI == 0) {
                    out[(size_t)gm * ldo + gn] = v;
                } else if (EPI == 1) {
                    const int bb = gm >> 11;   // batch = row / 2048
                    if (gn < INNER)
                        out[(size_t)gm * INNER + gn] = v * (1.f + aux1[bb * INNER + gn]);
                    else
                        out2[(size_t)gm * INNER + (gn - INNER)] = v + aux2[bb * INNER + (gn - INNER)];
                } else { // softplus
                    out[(size_t)gm * ldo + gn] = (v > 20.f) ? v : log1pf(__expf(v));
                }
            }
        }
}

// ---------------------------------------------------------------------------
// Kernel: depthwise causal conv (width 3, pad (2,0)) + SiLU
// ---------------------------------------------------------------------------
__global__ __launch_bounds__(256)
void conv_silu_kernel(const float* __restrict__ x,
                      const float* __restrict__ conv_w,
                      const float* __restrict__ conv_b,
                      float* __restrict__ out) {
    const size_t idx = (size_t)blockIdx.x * 256 + threadIdx.x;
    if (idx >= (size_t)NTOK * INNER) return;
    const int    c   = (int)(idx % INNER);
    const size_t row = idx / INNER;          // b*2048 + l
    const int    l   = (int)(row & (SEQL - 1));
    const float* w = conv_w + (size_t)c * 3;
    float acc = conv_b[c];
    if (l >= 2) acc += w[0] * x[(row - 2) * INNER + c];
    if (l >= 1) acc += w[1] * x[(row - 1) * INNER + c];
    acc += w[2] * x[row * INNER + c];
    out[idx] = siluf_(acc);
}

// ---------------------------------------------------------------------------
// Kernel: fused selective scan. One lane per (batch, channel); 16 states in
// registers; dA / dBx computed on the fly (never materialized).
// ---------------------------------------------------------------------------
__global__ __launch_bounds__(256)
void scan_kernel(const float* __restrict__ xconv,
                 const float* __restrict__ delta,
                 const float* __restrict__ xp,     // [NTOK, 80]: dt | B | C
                 const float* __restrict__ zbuf,
                 const float* __restrict__ log_A,  // [INNER, 16]
                 const float* __restrict__ Dvec,
                 float* __restrict__ gout) {
    const int tid = blockIdx.x * blockDim.x + threadIdx.x;
    if (tid >= 2 * INNER) return;
    const int b = tid / INNER;
    const int c = tid % INNER;

    float Ar[STATE], st[STATE];
#pragma unroll
    for (int s = 0; s < STATE; ++s) {
        Ar[s] = -__expf(log_A[(size_t)c * STATE + s]);
        st[s] = 0.f;
    }
    const float Dc = Dvec[c];

    for (int l = 0; l < SEQL; ++l) {
        const size_t row = (size_t)b * SEQL + l;
        const float dv = delta[row * INNER + c];
        const float xv = xconv[row * INNER + c];
        const float zv = zbuf[row * INNER + c];
        const float* xpr = xp + row * XPC;
        const float dx = dv * xv;
        float y = 0.f;
#pragma unroll
        for (int s = 0; s < STATE; ++s) {
            const float dA = __expf(dv * Ar[s]);
            st[s] = dA * st[s] + dx * xpr[DRANK + s];           // B_c
            y += st[s] * xpr[DRANK + STATE + s];                // C_c
        }
        y += xv * Dc;
        gout[row * INNER + c] = y * siluf_(zv);
    }
}

// ---------------------------------------------------------------------------
// Launch
// ---------------------------------------------------------------------------
extern "C" void kernel_launch(void* const* d_in, const int* in_sizes, int n_in,
                              void* d_out, int out_size, void* d_ws, size_t ws_size,
                              hipStream_t stream) {
    const float* tokens  = (const float*)d_in[0];
    const float* rctx    = (const float*)d_in[1];
    const float* W_in    = (const float*)d_in[2];
    const float* b_in    = (const float*)d_in[3];
    const float* conv_w  = (const float*)d_in[4];
    const float* conv_b  = (const float*)d_in[5];
    const float* W_x     = (const float*)d_in[6];
    const float* W_dt    = (const float*)d_in[7];
    const float* b_dt    = (const float*)d_in[8];
    const float* log_A   = (const float*)d_in[9];
    const float* Dvec    = (const float*)d_in[10];
    const float* W_out   = (const float*)d_in[11];
    const float* b_out   = (const float*)d_in[12];
    const float* ln_g    = (const float*)d_in[13];
    const float* ln_b    = (const float*)d_in[14];
    const float* W_scale = (const float*)d_in[15];
    const float* b_scale = (const float*)d_in[16];
    const float* W_gbias = (const float*)d_in[17];
    const float* b_gbias = (const float*)d_in[18];
    float* out = (float*)d_out;

    // ---- workspace layout ----  (float region, then ushort weight planes)
    float* ws     = (float*)d_ws;
    float* xbuf   = ws;                                 // also g after scan
    float* zbuf   = xbuf   + (size_t)NTOK * INNER;
    float* convb  = zbuf   + (size_t)NTOK * INNER;
    float* deltab = convb  + (size_t)NTOK * INNER;
    float* xpbuf  = deltab + (size_t)NTOK * INNER;
    float* drive  = xpbuf  + (size_t)NTOK * XPC;
    float* gbias  = drive  + 2 * INNER;

    unsigned short* winH  = (unsigned short*)(gbias + 2 * INNER);
    unsigned short* winL  = winH  + (size_t)3072 * 768;   // Np=3072, Kp=768
    unsigned short* woutH = winL  + (size_t)3072 * 768;
    unsigned short* woutL = woutH + (size_t)768 * 1536;   // Np=768, Kp=1536
    unsigned short* wxH   = woutL + (size_t)768 * 1536;
    unsigned short* wxL   = wxH   + (size_t)128 * 1536;   // Np=128, Kp=1536
    unsigned short* wdtH  = wxL   + (size_t)128 * 1536;
    unsigned short* wdtL  = wdtH  + (size_t)1536 * 64;    // Np=1536, Kp=64

    // 0) pre-split all weight matrices into transposed hi/lo f16 planes
    split_weight_kernel<<<(int)(((size_t)3072 * 768  + 255) / 256), 256, 0, stream>>>(
        W_in, 768, 3072, 768, 3072, winH, winL);
    split_weight_kernel<<<(int)(((size_t)768 * 1536  + 255) / 256), 256, 0, stream>>>(
        W_out, 1536, 768, 1536, 768, woutH, woutL);
    split_weight_kernel<<<(int)(((size_t)128 * 1536  + 255) / 256), 256, 0, stream>>>(
        W_x, 1536, 80, 1536, 128, wxH, wxL);
    split_weight_kernel<<<(int)(((size_t)1536 * 64   + 255) / 256), 256, 0, stream>>>(
        W_dt, 48, 1536, 64, 1536, wdtH, wdtL);

    // 1) context layernorm + drive/gbias
    ctx_kernel<<<2, 256, 0, stream>>>(rctx, ln_g, ln_b, W_scale, b_scale,
                                      W_gbias, b_gbias, drive, gbias);

    // 2) proj = tokens @ W_in + b_in  ->  x*(1+drive), z+gbias   (M=4096,N=3072,Kp=768)
    wmma_gemm_kernel<1><<<dim3(NTOK / 128, 3072 / 64), 256, 0, stream>>>(
        tokens, EMB, winH, winL, 768, b_in, xbuf, INNER, 2 * INNER,
        drive, gbias, zbuf);

    // 3) depthwise causal conv + SiLU
    conv_silu_kernel<<<(int)(((size_t)NTOK * INNER + 255) / 256), 256, 0, stream>>>(
        xbuf, conv_w, conv_b, convb);

    // 4) xp = x @ W_x   (M=4096, N=80 (Np=128), Kp=1536)
    wmma_gemm_kernel<0><<<dim3(NTOK / 128, 128 / 64), 256, 0, stream>>>(
        convb, INNER, wxH, wxL, 1536, nullptr, xpbuf, XPC, XPC,
        nullptr, nullptr, nullptr);

    // 5) delta = softplus(dt @ W_dt + b_dt)   (M=4096, N=1536, Kp=64; A = xp[:, :48],
    //    B rows 48..63 zero-padded so unguarded A reads are nullified)
    wmma_gemm_kernel<2><<<dim3(NTOK / 128, INNER / 64), 256, 0, stream>>>(
        xpbuf, XPC, wdtH, wdtL, 64, b_dt, deltab, INNER, INNER,
        nullptr, nullptr, nullptr);

    // 6) fused selective scan -> g = (scan(y) + x*D) * silu(z)  (into xbuf)
    scan_kernel<<<(2 * INNER + 255) / 256, 256, 0, stream>>>(
        convb, deltab, xpbuf, zbuf, log_A, Dvec, xbuf);

    // 7) out = g @ W_out + b_out   (M=4096, N=768, Kp=1536)
    wmma_gemm_kernel<0><<<dim3(NTOK / 128, EMB / 64), 256, 0, stream>>>(
        xbuf, INNER, woutH, woutL, 1536, b_out, out, EMB, EMB,
        nullptr, nullptr, nullptr);
}
